// AdvancedTransposableGene_9543417332457
// MI455X (gfx1250) — matrix operand order repaired and checked by reading, not compiled, not run on hardware
//
#include <hip/hip_runtime.h>
#include <hip/hip_bf16.h>

typedef __attribute__((ext_vector_type(16))) _Float16 v16h;
typedef __attribute__((ext_vector_type(8)))  float    v8f;

// ---------------- elementwise helpers ----------------

__global__ void set_val_kernel(float* __restrict__ p, long long n, float v) {
    long long i = (long long)blockIdx.x * blockDim.x + threadIdx.x;
    if (i < n) p[i] = v;
}

__global__ void deg_accum_kernel(const long long* __restrict__ dst,
                                 float* __restrict__ deg, int E) {
    int e = blockIdx.x * blockDim.x + threadIdx.x;
    if (e < E) atomicAdd(&deg[(int)dst[e]], 1.0f);
}

__global__ void rsqrt_inplace_kernel(float* __restrict__ p, int n) {
    int i = blockIdx.x * blockDim.x + threadIdx.x;
    if (i < n) {
        float d = p[i];
        p[i] = d > 0.0f ? rsqrtf(d) : 0.0f;
    }
}

// ---------------- weight pre-pack into WMMA B-fragment layout ----------------
// Target layout: P[((kt*TN + tn)*32 + lane)*16 + j] = (f16) W[k*NDIM + col]
//   where k = kt*32 + (lane>>4)*8 + (j&7) + (j>>3)*16, col = tn*16 + (lane&15).
// Each lane of the GEMM then loads its whole B fragment as 32 contiguous bytes.
template<int KDIM, int NDIM>
__global__ void pack_b_kernel(const float* __restrict__ W, _Float16* __restrict__ P) {
    const int TN = NDIM / 16;
    int idx = blockIdx.x * blockDim.x + threadIdx.x;
    if (idx >= KDIM * NDIM) return;
    int j    = idx & 15;
    int lane = (idx >> 4) & 31;
    int tile = idx >> 9;              // tile = kt*TN + tn
    int tn   = tile % TN;
    int kt   = tile / TN;
    int k    = kt * 32 + (lane >> 4) * 8 + (j & 7) + (j >> 3) * 16;
    int col  = tn * 16 + (lane & 15);
    P[idx] = (_Float16)W[(size_t)k * NDIM + col];
}

// ---------------- WMMA GEMM: C[M,NDIM] = A[M,KDIM] (f32->f16) @ W (pre-packed f16) ----------------
// One wave32 per 16-row block: A fragments loaded once into VGPRs, reused across
// all NDIM/16 column tiles. f32 accumulate; ADD_BIAS seeds C with bias[col].
template<int KDIM, int NDIM, bool ADD_BIAS>
__global__ void gemm_wmma_kernel(const float* __restrict__ A,
                                 const _Float16* __restrict__ Wp,
                                 const float* __restrict__ bias,
                                 float* __restrict__ C, int M) {
    const int TN = NDIM / 16;
    const int KT = KDIM / 32;
    int wave = blockIdx.x * (blockDim.x >> 5) + (threadIdx.x >> 5);
    int lane = threadIdx.x & 31;
    if (wave >= M / 16) return;        // wave-uniform guard: EXEC stays all-1s
    int l15  = lane & 15;
    int half = lane >> 4;              // 0: lanes 0-15, 1: lanes 16-31
    int row  = wave * 16 + l15;        // A row held by this lane

    // Load all A fragments for this 16-row block (ISA 16-bit A layout):
    // lane holds row `row`; K = ka..ka+7 and ka+16..ka+23, ka = kt*32 + half*8.
    v16h afrag[KT];
    const float* arow = A + (size_t)row * KDIM;
    #pragma unroll
    for (int kt = 0; kt < KT; ++kt) {
        int ka = kt * 32 + half * 8;
        float4 p0 = *(const float4*)(arow + ka);
        float4 p1 = *(const float4*)(arow + ka + 4);
        float4 p2 = *(const float4*)(arow + ka + 16);
        float4 p3 = *(const float4*)(arow + ka + 20);
        v16h a;
        a[0]  = (_Float16)p0.x; a[1]  = (_Float16)p0.y;
        a[2]  = (_Float16)p0.z; a[3]  = (_Float16)p0.w;
        a[4]  = (_Float16)p1.x; a[5]  = (_Float16)p1.y;
        a[6]  = (_Float16)p1.z; a[7]  = (_Float16)p1.w;
        a[8]  = (_Float16)p2.x; a[9]  = (_Float16)p2.y;
        a[10] = (_Float16)p2.z; a[11] = (_Float16)p2.w;
        a[12] = (_Float16)p3.x; a[13] = (_Float16)p3.y;
        a[14] = (_Float16)p3.z; a[15] = (_Float16)p3.w;
        afrag[kt] = a;
    }

    int m0 = wave * 16 + half * 8;     // C rows held in VGPRs r=0..7
    #pragma unroll
    for (int tn = 0; tn < TN; ++tn) {
        int col = tn * 16 + l15;
        v8f c = {};
        if (ADD_BIAS) {
            float bv = bias[col];
            #pragma unroll
            for (int r = 0; r < 8; ++r) c[r] = bv;
        }
        #pragma unroll
        for (int kt = 0; kt < KT; ++kt) {
            // B fragment: 32 contiguous bytes per lane (pre-packed)
            v16h b = *(const v16h*)(Wp + (((size_t)kt * TN + tn) * 32 + lane) * 16);
            c = __builtin_amdgcn_wmma_f32_16x16x32_f16(
                    /*neg_a=*/false, afrag[kt], /*neg_b=*/false, b,
                    /*c_mod=*/(short)0, c, /*reuse_a=*/false, /*reuse_b=*/false);
        }
        #pragma unroll
        for (int r = 0; r < 8; ++r)
            C[(size_t)(m0 + r) * NDIM + col] = c[r];
    }
}

// ---------------- edge scatter: agg[dst] += dis[src]*dis[dst] * h[src] ----------------
template<int F>
__global__ void edge_scatter_kernel(const long long* __restrict__ src,
                                    const long long* __restrict__ dst,
                                    const float* __restrict__ dis,
                                    const float* __restrict__ h,
                                    float* __restrict__ agg, int E) {
    const int CH = F / 4;                       // float4 chunks per edge
    long long idx = (long long)blockIdx.x * blockDim.x + threadIdx.x;
    if (idx >= (long long)E * CH) return;
    int e  = (int)(idx / CH);
    int f4 = (int)(idx % CH) * 4;
    int s = (int)src[e];
    int d = (int)dst[e];
    float nrm = dis[s] * dis[d];
    const float4 hv = *(const float4*)(h + (size_t)s * F + f4);
    float* o = agg + (size_t)d * F + f4;
    atomicAdd(o + 0, nrm * hv.x);
    atomicAdd(o + 1, nrm * hv.y);
    atomicAdd(o + 2, nrm * hv.z);
    atomicAdd(o + 3, nrm * hv.w);
}

// ---------------- finalize: agg = act(agg + dis[i]^2 * h + bias[f]) ----------------
template<int F, bool RELU>
__global__ void finalize_kernel(const float* __restrict__ dis,
                                const float* __restrict__ h,
                                const float* __restrict__ bias,
                                float* __restrict__ agg, int N) {
    long long idx = (long long)blockIdx.x * blockDim.x + threadIdx.x;
    if (idx >= (long long)N * F) return;
    int i = (int)(idx / F);
    int f = (int)(idx % F);
    float di = dis[i];
    float v = agg[idx] + di * di * h[idx] + bias[f];
    agg[idx] = RELU ? fmaxf(v, 0.0f) : v;
}

// ---------------- pooling ----------------

__global__ void pool_count_kernel(const long long* __restrict__ batch,
                                  float* __restrict__ cnt, int N) {
    int i = blockIdx.x * blockDim.x + threadIdx.x;
    if (i < N) atomicAdd(&cnt[(int)batch[i]], 1.0f);
}

template<int F>
__global__ void pool_sum_kernel(const long long* __restrict__ batch,
                                const float* __restrict__ h,
                                float* __restrict__ sums, int N) {
    long long idx = (long long)blockIdx.x * blockDim.x + threadIdx.x;
    if (idx >= (long long)N * F) return;
    int i = (int)(idx / F);
    int f = (int)(idx % F);
    atomicAdd(&sums[(int)batch[i] * F + f], h[idx]);
}

template<int F>
__global__ void pool_final_kernel(const float* __restrict__ sums,
                                  const float* __restrict__ cnt,
                                  float* __restrict__ out, int G) {
    int idx = blockIdx.x * blockDim.x + threadIdx.x;
    if (idx >= G * F) return;
    out[idx] = sums[idx] / fmaxf(cnt[idx / F], 1.0f);
}

// ---------------- launch ----------------

extern "C" void kernel_launch(void* const* d_in, const int* in_sizes, int n_in,
                              void* d_out, int out_size, void* d_ws, size_t ws_size,
                              hipStream_t stream) {
    const int N = 100000, E = 1000000, G = 64;
    const int F_IN = 32, H2 = 128, H = 64;

    const float*     x     = (const float*)d_in[0];
    const long long* ei    = (const long long*)d_in[1];
    const long long* batch = (const long long*)d_in[2];
    const float*     W1    = (const float*)d_in[3];
    const float*     b1    = (const float*)d_in[4];
    const float*     W2    = (const float*)d_in[5];
    const float*     b2    = (const float*)d_in[6];
    const float*     Wrt   = (const float*)d_in[7];
    const float*     brt   = (const float*)d_in[8];
    float*           out   = (float*)d_out;

    const long long* srcE = ei;          // edge_index[0]
    const long long* dstE = ei + E;      // edge_index[1]

    // Workspace carve-up (all sub-buffers 32B-aligned; with reuse):
    float* ws   = (float*)d_ws;
    float* dis  = ws;                               // [N]      degree -> deg^{-1/2}
    float* h1   = dis  + N;                         // [N,128]  h1; reused as agg2/a2 ([N,64])
    float* agg1 = h1   + (size_t)N * H2;            // [N,128]  agg1 -> a1 (in-place)
    float* h2   = agg1 + (size_t)N * H2;            // [N,64]   h2; reused as h3
    float* psum = h2   + (size_t)N * H;             // [G,64]
    float* pcnt = psum + (size_t)G * H;             // [G]
    _Float16* w1p = (_Float16*)(pcnt + G);          // [32,128] packed B fragments
    _Float16* w2p = w1p + F_IN * H2;                // [128,64]
    _Float16* wrp = w2p + H2 * H;                   // [64,64]

    const int B = 256;
    auto blocks = [](long long n) { return (unsigned)((n + 255) / 256); };

    // --- degree + normalization (self-loop contributes 1 to every node) ---
    set_val_kernel<<<blocks(N), B, 0, stream>>>(dis, N, 1.0f);
    deg_accum_kernel<<<blocks(E), B, 0, stream>>>(dstE, dis, E);
    rsqrt_inplace_kernel<<<blocks(N), B, 0, stream>>>(dis, N);

    // --- pack weights into f16 WMMA B-fragment layout (one-time, tiny) ---
    pack_b_kernel<F_IN, H2><<<blocks(F_IN * H2), B, 0, stream>>>(W1,  w1p);
    pack_b_kernel<H2, H>   <<<blocks(H2 * H),    B, 0, stream>>>(W2,  w2p);
    pack_b_kernel<H, H>    <<<blocks(H * H),     B, 0, stream>>>(Wrt, wrp);

    // --- layer 1: h1 = x @ W1 (WMMA), aggregate, relu(+b1) ---
    gemm_wmma_kernel<F_IN, H2, false><<<blocks((long long)(N / 16) * 32), B, 0, stream>>>(x, w1p, nullptr, h1, N);
    set_val_kernel<<<blocks((long long)N * H2), B, 0, stream>>>(agg1, (long long)N * H2, 0.0f);
    edge_scatter_kernel<H2><<<blocks((long long)E * (H2 / 4)), B, 0, stream>>>(srcE, dstE, dis, h1, agg1, E);
    finalize_kernel<H2, true><<<blocks((long long)N * H2), B, 0, stream>>>(dis, h1, b1, agg1, N);

    // --- layer 2: h2 = a1 @ W2 (WMMA), aggregate (into h1's region), +b2 ---
    gemm_wmma_kernel<H2, H, false><<<blocks((long long)(N / 16) * 32), B, 0, stream>>>(agg1, w2p, nullptr, h2, N);
    float* agg2 = h1;   // h1 dead after finalize1
    set_val_kernel<<<blocks((long long)N * H), B, 0, stream>>>(agg2, (long long)N * H, 0.0f);
    edge_scatter_kernel<H><<<blocks((long long)E * (H / 4)), B, 0, stream>>>(srcE, dstE, dis, h2, agg2, E);
    finalize_kernel<H, false><<<blocks((long long)N * H), B, 0, stream>>>(dis, h2, b2, agg2, N);

    // --- h3 = a2 @ Wrt + brt (WMMA, bias seeded into accumulator); h3 overwrites h2 ---
    float* h3 = h2;     // h2 dead after finalize2
    gemm_wmma_kernel<H, H, true><<<blocks((long long)(N / 16) * 32), B, 0, stream>>>(agg2, wrp, brt, h3, N);

    // --- global mean pool ---
    set_val_kernel<<<blocks(G * H + G), B, 0, stream>>>(psum, G * H + G, 0.0f);
    pool_count_kernel<<<blocks(N), B, 0, stream>>>(batch, pcnt, N);
    pool_sum_kernel<H><<<blocks((long long)N * H), B, 0, stream>>>(batch, h3, psum, N);
    pool_final_kernel<H><<<blocks(G * H), B, 0, stream>>>(psum, pcnt, out, G);
}
